// FusedEncoderBlock_54236847014070
// MI455X (gfx1250) — matrix operand order repaired and checked by reading, not compiled, not run on hardware
//
#include <hip/hip_runtime.h>
#include <hip/hip_bf16.h>
#include <math.h>

// ---------------- types ----------------
typedef __attribute__((ext_vector_type(16))) __bf16        v16bf;
typedef __attribute__((ext_vector_type(8)))  float         v8f;
typedef __attribute__((ext_vector_type(4)))  unsigned int  u32x4;

static __device__ __forceinline__ v16bf mkfrag(u32x4 lo, u32x4 hi) {
  union { u32x4 q[2]; v16bf v; } u;
  u.q[0] = lo; u.q[1] = hi;
  return u.v;
}

static __device__ __forceinline__ v8f wmma_bf16(v16bf a, v16bf b, v8f c) {
  return __builtin_amdgcn_wmma_f32_16x16x32_bf16(false, a, false, b, (short)0, c,
                                                 false, false);
}

// ---------------- constants ----------------
#define S_LEN 2048
#define DMODEL 1024
#define NHEADS 16
#define DHEAD 64
#define FFDIM 4096

// workspace layout (bytes)
#define OFF_WQKV   (size_t)0            //  6 MB bf16 1024x3072
#define OFF_WATTN  (size_t)6291456      //  2 MB bf16 1024x1024
#define OFF_WFF1   (size_t)8388608      //  8 MB bf16 1024x4096
#define OFF_WFF2   (size_t)16777216     //  8 MB bf16 4096x1024
#define OFF_H      (size_t)25165824     //  4 MB bf16 2048x1024  (ln1 out / ln2 out)
#define OFF_QKV    (size_t)29360128     // 12 MB bf16 2048x3072
#define OFF_QR     (size_t)41943040     //  4 MB bf16 2048x1024
#define OFF_KR     (size_t)46137344     //  4 MB bf16 2048x1024
#define OFF_ATT    (size_t)50331648     //  4 MB bf16 2048x1024
#define OFF_XMID   (size_t)54525952     //  8 MB f32  2048x1024
#define OFF_FF1    OFF_QKV              // 16 MB bf16 2048x4096 (aliases QKV+QR, dead by then)

// ---------------- fp32 -> bf16 convert ----------------
__global__ __launch_bounds__(256) void cvt_bf16_kernel(const float* __restrict__ in,
                                                       __bf16* __restrict__ out, int n) {
  int i = blockIdx.x * 256 + threadIdx.x;
  if (i < n) out[i] = (__bf16)in[i];
}

// ---------------- layernorm (one row per block) ----------------
__global__ __launch_bounds__(256) void ln_kernel(const float* __restrict__ x,
                                                 const float* __restrict__ g,
                                                 const float* __restrict__ b,
                                                 __bf16* __restrict__ out) {
  __shared__ float r1[256];
  __shared__ float r2[256];
  const int row = blockIdx.x;
  const int tid = threadIdx.x;
  const float* xr = x + (size_t)row * DMODEL;
  float s = 0.f, s2 = 0.f;
  float v[4];
#pragma unroll
  for (int i = 0; i < 4; ++i) {
    v[i] = xr[tid + i * 256];
    s += v[i];
    s2 += v[i] * v[i];
  }
  r1[tid] = s; r2[tid] = s2;
  __syncthreads();
  for (int off = 128; off > 0; off >>= 1) {
    if (tid < off) { r1[tid] += r1[tid + off]; r2[tid] += r2[tid + off]; }
    __syncthreads();
  }
  const float mu  = r1[0] * (1.0f / DMODEL);
  const float var = r2[0] * (1.0f / DMODEL) - mu * mu;
  const float inv = rsqrtf(var + 1e-5f);
  __bf16* orow = out + (size_t)row * DMODEL;
#pragma unroll
  for (int i = 0; i < 4; ++i) {
    int c = tid + i * 256;
    orow[c] = (__bf16)((v[i] - mu) * inv * g[c] + b[c]);
  }
}

// ---------------- RoPE on q,k from packed qkv ----------------
__global__ __launch_bounds__(256) void rope_kernel(const __bf16* __restrict__ qkv,
                                                   __bf16* __restrict__ qr,
                                                   __bf16* __restrict__ kr) {
  int t = blockIdx.x * 256 + threadIdx.x;
  if (t >= S_LEN * NHEADS * 32) return;
  const int s   = t >> 9;
  const int rem = t & 511;
  const int h   = rem >> 5;
  const int i   = rem & 31;
  const size_t base = (size_t)s * (3 * DMODEL) + h * DHEAD + i;
  const float q1 = (float)qkv[base];
  const float q2 = (float)qkv[base + 32];
  const float k1 = (float)qkv[base + DMODEL];
  const float k2 = (float)qkv[base + DMODEL + 32];
  const float invf = __expf(-0.28782313662425572f * (float)i); // 10000^(-i/32)
  const float ang = (float)s * invf;
  float sn, cs;
  __sincosf(ang, &sn, &cs);
  const size_t ob = (size_t)s * DMODEL + h * DHEAD + i;
  qr[ob]      = (__bf16)(q1 * cs - q2 * sn);
  qr[ob + 32] = (__bf16)(q2 * cs + q1 * sn);
  kr[ob]      = (__bf16)(k1 * cs - k2 * sn);
  kr[ob + 32] = (__bf16)(k2 * cs + k1 * sn);
}

// ---------------- generic WMMA GEMM: C = A(MxK) * B(KxN) [+bias][gelu][+resid] ----------------
// block tile 128(M) x 128(N), 8 waves (4 along M x 2 along N), each wave 32x64 via 2x4 WMMA
#define LDA_PAD 40   // bf16 elems per LDS row (80B, 16B aligned)
template <bool BIAS, bool GELU_ACT, bool RESID, bool OUTF32>
__global__ __launch_bounds__(256) void gemm_bf16_kernel(
    const __bf16* __restrict__ A, const __bf16* __restrict__ B,
    const float* __restrict__ bias, const float* __restrict__ resid,
    void* __restrict__ out, int M, int N, int K) {
  __shared__ __bf16 smA[128 * LDA_PAD];
  __shared__ __bf16 smB[128 * LDA_PAD];   // transposed: [n][kk]
  const int tid  = threadIdx.x;
  const int wave = tid >> 5;
  const int lane = tid & 31;
  const int half = lane >> 4;
  const int l16  = lane & 15;
  const int wm = (wave & 3) * 32;   // 4 waves along M
  const int wn = (wave >> 2) * 64;  // 2 waves along N
  const int n0 = blockIdx.x * 128;
  const int m0 = blockIdx.y * 128;

  v8f acc[2][4];
#pragma unroll
  for (int t = 0; t < 2; ++t)
#pragma unroll
    for (int u = 0; u < 4; ++u) acc[t][u] = (v8f){0.f,0.f,0.f,0.f,0.f,0.f,0.f,0.f};

  for (int k0 = 0; k0 < K; k0 += 32) {
    // A tile 128x32: 512 16B-chunks, 2 per thread (vector load + vector LDS store)
#pragma unroll
    for (int i = 0; i < 2; ++i) {
      int chunk = tid + i * 256;
      int m = chunk >> 2, part = chunk & 3;
      u32x4 v = *reinterpret_cast<const u32x4*>(A + (size_t)(m0 + m) * K + k0 + part * 8);
      *reinterpret_cast<u32x4*>(&smA[m * LDA_PAD + part * 8]) = v;
    }
    // B tile 32x128: 512 16B-chunks (8 n-values each), transpose-scatter into LDS
#pragma unroll
    for (int i = 0; i < 2; ++i) {
      int chunk = tid + i * 256;
      int kk = chunk >> 4, nc = chunk & 15;   // nc*8 = n offset
      union { u32x4 q; __bf16 e[8]; } bu;
      bu.q = *reinterpret_cast<const u32x4*>(B + (size_t)(k0 + kk) * N + n0 + nc * 8);
#pragma unroll
      for (int j = 0; j < 8; ++j) smB[(nc * 8 + j) * LDA_PAD + kk] = bu.e[j];
    }
    __syncthreads();

    v16bf afr[2], bfr[4];
#pragma unroll
    for (int t = 0; t < 2; ++t) {
      int row = wm + t * 16 + l16;
      const __bf16* p = &smA[row * LDA_PAD];
      afr[t] = mkfrag(*reinterpret_cast<const u32x4*>(p + half * 8),
                      *reinterpret_cast<const u32x4*>(p + 16 + half * 8));
    }
#pragma unroll
    for (int u = 0; u < 4; ++u) {
      int col = wn + u * 16 + l16;
      const __bf16* p = &smB[col * LDA_PAD];
      bfr[u] = mkfrag(*reinterpret_cast<const u32x4*>(p + half * 8),
                      *reinterpret_cast<const u32x4*>(p + 16 + half * 8));
    }
#pragma unroll
    for (int t = 0; t < 2; ++t)
#pragma unroll
      for (int u = 0; u < 4; ++u) acc[t][u] = wmma_bf16(afr[t], bfr[u], acc[t][u]);
    __syncthreads();
  }

  // epilogue
#pragma unroll
  for (int u = 0; u < 4; ++u) {
    const int col = n0 + wn + u * 16 + l16;
    float bv = 0.f;
    if (BIAS) bv = bias[col];
#pragma unroll
    for (int t = 0; t < 2; ++t) {
#pragma unroll
      for (int r = 0; r < 8; ++r) {
        const int row = m0 + wm + t * 16 + r + half * 8;
        float v = acc[t][u][r] + bv;
        if (GELU_ACT) v = 0.5f * v * (1.0f + erff(v * 0.70710678118654752f));
        if (RESID) v += resid[(size_t)row * N + col];
        if (OUTF32) ((float*)out)[(size_t)row * N + col] = v;
        else        ((__bf16*)out)[(size_t)row * N + col] = (__bf16)v;
      }
    }
  }
}

// ---------------- flash attention (causal), one WG = (head, 128 query rows) ----------------
#define KSTR 72   // LDS stride (144B, 16B aligned)
__global__ __launch_bounds__(256) void attn_kernel(const __bf16* __restrict__ qr,
                                                   const __bf16* __restrict__ kr,
                                                   const __bf16* __restrict__ qkv,
                                                   __bf16* __restrict__ att) {
  __shared__ __bf16 smK[64 * KSTR];    // [key][d]
  __shared__ __bf16 smVT[64 * KSTR];   // [d][key]
  __shared__ __bf16 smP[8 * 16 * KSTR];
  const int tid  = threadIdx.x;
  const int wave = tid >> 5;
  const int lane = tid & 31;
  const int half = lane >> 4;
  const int l16  = lane & 15;
  const int h  = blockIdx.y;
  const int q0 = blockIdx.x * 128;
  const int qw = q0 + wave * 16;   // this wave's first query row

  // load Q fragments once (A layout: row = lane&15, interleaved K halves)
  v16bf qf[2];
#pragma unroll
  for (int kc = 0; kc < 2; ++kc) {
    const __bf16* p = qr + (size_t)(qw + l16) * DMODEL + h * DHEAD + kc * 32;
    qf[kc] = mkfrag(*reinterpret_cast<const u32x4*>(p + half * 8),
                    *reinterpret_cast<const u32x4*>(p + 16 + half * 8));
  }

  v8f oacc[4];
#pragma unroll
  for (int u = 0; u < 4; ++u) oacc[u] = (v8f){0.f,0.f,0.f,0.f,0.f,0.f,0.f,0.f};
  float mrow[8], lrow[8];
#pragma unroll
  for (int r = 0; r < 8; ++r) { mrow[r] = -3.0e38f; lrow[r] = 0.f; }

  __bf16* pb = &smP[wave * 16 * KSTR];

  for (int kb = 0; kb < q0 + 128; kb += 64) {
    // stage K block (vector) and V^T block (vector load, transpose-scatter)
#pragma unroll
    for (int i = 0; i < 2; ++i) {
      int chunk = tid + i * 256;           // 512 chunks of 8 d-values
      int key = chunk >> 3, dc = chunk & 7;
      int d0 = dc * 8;
      u32x4 kv = *reinterpret_cast<const u32x4*>(
          kr + (size_t)(kb + key) * DMODEL + h * DHEAD + d0);
      *reinterpret_cast<u32x4*>(&smK[key * KSTR + d0]) = kv;
      union { u32x4 q; __bf16 e[8]; } vu;
      vu.q = *reinterpret_cast<const u32x4*>(
          qkv + (size_t)(kb + key) * (3 * DMODEL) + 2 * DMODEL + h * DHEAD + d0);
#pragma unroll
      for (int j = 0; j < 8; ++j) smVT[(d0 + j) * KSTR + key] = vu.e[j];
    }
    __syncthreads();

    // S = Q K^T  (4 key tiles of 16)
    v8f sacc[4];
#pragma unroll
    for (int u = 0; u < 4; ++u) sacc[u] = (v8f){0.f,0.f,0.f,0.f,0.f,0.f,0.f,0.f};
#pragma unroll
    for (int u = 0; u < 4; ++u) {
#pragma unroll
      for (int kc = 0; kc < 2; ++kc) {
        const __bf16* p = &smK[(u * 16 + l16) * KSTR + kc * 32];
        v16bf bfr = mkfrag(*reinterpret_cast<const u32x4*>(p + half * 8),
                           *reinterpret_cast<const u32x4*>(p + 16 + half * 8));
        sacc[u] = wmma_bf16(qf[kc], bfr, sacc[u]);
      }
    }
    // scale + causal mask  (C layout: row = r + 8*half, col = u*16 + l16)
#pragma unroll
    for (int u = 0; u < 4; ++u) {
      int key = kb + u * 16 + l16;
#pragma unroll
      for (int r = 0; r < 8; ++r) {
        int row = qw + r + half * 8;
        float s = sacc[u][r] * 0.125f;
        if (key > row) s = -3.0e38f;
        sacc[u][r] = s;
      }
    }
    // online softmax
    float mnew[8];
#pragma unroll
    for (int r = 0; r < 8; ++r) {
      float t = fmaxf(fmaxf(sacc[0][r], sacc[1][r]), fmaxf(sacc[2][r], sacc[3][r]));
#pragma unroll
      for (int m = 1; m < 16; m <<= 1) t = fmaxf(t, __shfl_xor(t, m, 32));
      mnew[r] = fmaxf(mrow[r], t);
      float alpha = __expf(mrow[r] - mnew[r]);
      lrow[r] *= alpha;
#pragma unroll
      for (int u = 0; u < 4; ++u) oacc[u][r] *= alpha;
    }
    float rsum[8];
#pragma unroll
    for (int r = 0; r < 8; ++r) rsum[r] = 0.f;
#pragma unroll
    for (int u = 0; u < 4; ++u)
#pragma unroll
      for (int r = 0; r < 8; ++r) {
        float p = __expf(sacc[u][r] - mnew[r]);
        sacc[u][r] = p;
        rsum[r] += p;
      }
#pragma unroll
    for (int r = 0; r < 8; ++r) {
#pragma unroll
      for (int m = 1; m < 16; m <<= 1) rsum[r] += __shfl_xor(rsum[r], m, 32);
      lrow[r] += rsum[r];
      mrow[r] = mnew[r];
    }
    // transpose P through wave-local LDS into A-fragment layout
#pragma unroll
    for (int u = 0; u < 4; ++u)
#pragma unroll
      for (int r = 0; r < 8; ++r)
        pb[(r + half * 8) * KSTR + u * 16 + l16] = (__bf16)sacc[u][r];

    v16bf pf[2];
#pragma unroll
    for (int kc = 0; kc < 2; ++kc) {
      const __bf16* p = pb + l16 * KSTR + kc * 32;
      pf[kc] = mkfrag(*reinterpret_cast<const u32x4*>(p + half * 8),
                      *reinterpret_cast<const u32x4*>(p + 16 + half * 8));
    }
    // O += P V
#pragma unroll
    for (int u = 0; u < 4; ++u) {
#pragma unroll
      for (int kc = 0; kc < 2; ++kc) {
        const __bf16* p = &smVT[(u * 16 + l16) * KSTR + kc * 32];
        v16bf bfr = mkfrag(*reinterpret_cast<const u32x4*>(p + half * 8),
                           *reinterpret_cast<const u32x4*>(p + 16 + half * 8));
        oacc[u] = wmma_bf16(pf[kc], bfr, oacc[u]);
      }
    }
    __syncthreads();
  }

  // normalize + store
#pragma unroll
  for (int u = 0; u < 4; ++u) {
#pragma unroll
    for (int r = 0; r < 8; ++r) {
      int row = qw + r + half * 8;
      int d = h * DHEAD + u * 16 + l16;
      att[(size_t)row * DMODEL + d] = (__bf16)(oacc[u][r] / lrow[r]);
    }
  }
}

// ---------------- launcher ----------------
extern "C" void kernel_launch(void* const* d_in, const int* in_sizes, int n_in,
                              void* d_out, int out_size, void* d_ws, size_t ws_size,
                              hipStream_t stream) {
  const float* x      = (const float*)d_in[0];
  const float* w_qkv  = (const float*)d_in[1];
  const float* w_attn = (const float*)d_in[2];
  const float* b_attn = (const float*)d_in[3];
  const float* w_ff1  = (const float*)d_in[4];
  const float* b_ff1  = (const float*)d_in[5];
  const float* w_ff2  = (const float*)d_in[6];
  const float* b_ff2  = (const float*)d_in[7];
  const float* ln1_g  = (const float*)d_in[8];
  const float* ln1_b  = (const float*)d_in[9];
  const float* ln2_g  = (const float*)d_in[10];
  const float* ln2_b  = (const float*)d_in[11];
  float* out = (float*)d_out;

  char* ws = (char*)d_ws;
  __bf16* wqkv_b  = (__bf16*)(ws + OFF_WQKV);
  __bf16* wattn_b = (__bf16*)(ws + OFF_WATTN);
  __bf16* wff1_b  = (__bf16*)(ws + OFF_WFF1);
  __bf16* wff2_b  = (__bf16*)(ws + OFF_WFF2);
  __bf16* h_b     = (__bf16*)(ws + OFF_H);
  __bf16* qkv_b   = (__bf16*)(ws + OFF_QKV);
  __bf16* qr_b    = (__bf16*)(ws + OFF_QR);
  __bf16* kr_b    = (__bf16*)(ws + OFF_KR);
  __bf16* att_b   = (__bf16*)(ws + OFF_ATT);
  float*  xmid_f  = (float*)(ws + OFF_XMID);
  __bf16* ff1_b   = (__bf16*)(ws + OFF_FF1);

  // weights -> bf16
  cvt_bf16_kernel<<<(DMODEL * 3 * DMODEL + 255) / 256, 256, 0, stream>>>(w_qkv, wqkv_b, DMODEL * 3 * DMODEL);
  cvt_bf16_kernel<<<(DMODEL * DMODEL + 255) / 256, 256, 0, stream>>>(w_attn, wattn_b, DMODEL * DMODEL);
  cvt_bf16_kernel<<<(DMODEL * FFDIM + 255) / 256, 256, 0, stream>>>(w_ff1, wff1_b, DMODEL * FFDIM);
  cvt_bf16_kernel<<<(FFDIM * DMODEL + 255) / 256, 256, 0, stream>>>(w_ff2, wff2_b, FFDIM * DMODEL);

  // LN1
  ln_kernel<<<S_LEN, 256, 0, stream>>>(x, ln1_g, ln1_b, h_b);

  // qkv = h @ w_qkv  (2048x1024 @ 1024x3072)
  gemm_bf16_kernel<false, false, false, false>
      <<<dim3(3 * DMODEL / 128, S_LEN / 128), 256, 0, stream>>>(
          h_b, wqkv_b, nullptr, nullptr, qkv_b, S_LEN, 3 * DMODEL, DMODEL);

  // RoPE
  rope_kernel<<<(S_LEN * NHEADS * 32) / 256, 256, 0, stream>>>(qkv_b, qr_b, kr_b);

  // attention
  attn_kernel<<<dim3(S_LEN / 128, NHEADS), 256, 0, stream>>>(qr_b, kr_b, qkv_b, att_b);

  // x_mid = x + att @ w_attn + b_attn
  gemm_bf16_kernel<true, false, true, true>
      <<<dim3(DMODEL / 128, S_LEN / 128), 256, 0, stream>>>(
          att_b, wattn_b, b_attn, x, xmid_f, S_LEN, DMODEL, DMODEL);

  // LN2
  ln_kernel<<<S_LEN, 256, 0, stream>>>(xmid_f, ln2_g, ln2_b, h_b);

  // ff1 = gelu(h2 @ w_ff1 + b_ff1)
  gemm_bf16_kernel<true, true, false, false>
      <<<dim3(FFDIM / 128, S_LEN / 128), 256, 0, stream>>>(
          h_b, wff1_b, b_ff1, nullptr, ff1_b, S_LEN, FFDIM, DMODEL);

  // out = x_mid + ff1 @ w_ff2 + b_ff2
  gemm_bf16_kernel<true, false, true, true>
      <<<dim3(DMODEL / 128, S_LEN / 128), 256, 0, stream>>>(
          ff1_b, wff2_b, b_ff2, xmid_f, out, S_LEN, DMODEL, FFDIM);
}
